// gateway_sw_6897717477459
// MI455X (gfx1250) — compile-verified, matrix operand order checked by reading, not run on hardware
//
#include <hip/hip_runtime.h>
#include <math.h>

// Model dims
#define NATT 11
#define BB   8
#define SS   512
#define DD   768
#define HH   256
#define H3   768            // 3*H
#define NSEQ (NATT * BB)    // 88
#define MQKV (BB * SS)      // 4096
#define MXP  (NSEQ * SS)    // 45056

typedef __attribute__((ext_vector_type(16))) __bf16 v16bf;
typedef __attribute__((ext_vector_type(8)))  float  v8f;

__device__ __forceinline__ __bf16 f2bf(float f) {
  unsigned u; __builtin_memcpy(&u, &f, 4);
  unsigned r = u + 0x7FFFu + ((u >> 16) & 1u);   // round-to-nearest-even
  unsigned short h = (unsigned short)(r >> 16);
  __bf16 o; __builtin_memcpy(&o, &h, 2); return o;
}
__device__ __forceinline__ float bf2f(__bf16 x) {
  unsigned short s; __builtin_memcpy(&s, &x, 2);
  unsigned u = ((unsigned)s) << 16; float f; __builtin_memcpy(&f, &u, 4); return f;
}

// K offset inside a 16-bit 16x32 A/B WMMA fragment (ISA 7.12.2):
// lanes 0-15 read K = {0..7,16..23}; lanes 16-31 read K = {8..15,24..31}.
__device__ __forceinline__ int frag_koff(int lane, int e) {
  int p = e >> 1;
  return ((lane & 16) ? 8 : 0) + ((p & 4) ? (8 + 2 * p) : (2 * p)) + (e & 1);
}
// Fragment-order permutation within a 32-element K chunk = swap bits 3 and 4.
// Maps aligned 8-element blocks to aligned 8-element blocks -> 16B global
// chunks land at 16B-aligned LDS addresses (async-to-LDS friendly).
__device__ __forceinline__ int kperm(int k) {
  return (k & 7) | ((k & 8) << 1) | ((k & 16) >> 1);
}

// 32-byte fragment load as two 16B chunks (ds_load_b128 / global_load_b128).
__device__ __forceinline__ v16bf load_frag(const __bf16* p) {
  uint4 lo = *(const uint4*)(p);
  uint4 hi = *(const uint4*)(p + 8);
  v16bf r;
  __builtin_memcpy(&r, &lo, 16);
  __builtin_memcpy((char*)&r + 16, &hi, 16);
  return r;
}

// gfx1250 async global->LDS copy (ASYNCcnt tracked).
__device__ __forceinline__ void async_copy_b128(void* lds_dst, const void* gsrc) {
  unsigned off = (unsigned)(size_t)lds_dst;
  asm volatile("global_load_async_to_lds_b128 %0, %1, off"
               :: "v"(off), "v"(gsrc) : "memory");
}
__device__ __forceinline__ void async_wait0() {
  asm volatile("s_wait_asynccnt 0x0" ::: "memory");
}

#define LDA_P 72    // A tile pitch (elements); 144B rows, 16B aligned
#define LDB_P 40    // B tile pitch (elements); stored [n][kperm(k)], 80B rows

// 128x128 block-tile GEMM core. 256 threads = 8 waves; each wave owns a 64x32
// sub-tile (4 M-frags x 2 N-frags). KTILE=32. Tiles live in LDS in WMMA
// fragment order; staging uses async-to-LDS b128 (A always; B when TRANSB).
// TRANSB=false: Bsrc row-major [K][ldbs];  TRANSB=true: Bsrc row-major [N][ldbs].
template <bool TRANSB>
__device__ __forceinline__ void gemm_block_128x128(
    const __bf16* __restrict__ A, int lda, int m0,
    const __bf16* __restrict__ Bsrc, int ldbs, int n0,
    int Kdim, __bf16* sA, __bf16* sB, v8f acc[4][2]) {
  const int tid  = threadIdx.x;
  const int lane = tid & 31;
  const int wave = tid >> 5;
  const int wm   = wave & 1;
  const int wn   = wave >> 1;

  for (int kt = 0; kt < Kdim; kt += 32) {
    // ---- stage A: 128 rows x 4 chunks of 8 bf16, async b128 deposits ----
    #pragma unroll
    for (int u = tid; u < 512; u += 256) {
      int row = u >> 2, c0 = (u & 3) << 3;
      async_copy_b128(sA + row * LDA_P + kperm(c0),
                      A + (size_t)(m0 + row) * lda + kt + c0);
    }
    // ---- stage B into [n][kperm(k)] ----
    if (TRANSB) {
      #pragma unroll
      for (int u = tid; u < 512; u += 256) {
        int n = u >> 2, c0 = (u & 3) << 3;
        async_copy_b128(sB + n * LDB_P + kperm(c0),
                        Bsrc + (size_t)(n0 + n) * ldbs + kt + c0);
      }
    } else {
      #pragma unroll
      for (int u = tid; u < 512; u += 256) {
        int k = u >> 4, c0 = (u & 15) << 3;
        uint4 val = *(const uint4*)(Bsrc + (size_t)(kt + k) * ldbs + n0 + c0);
        __bf16 tmp[8];
        __builtin_memcpy(tmp, &val, 16);
        int kp = kperm(k);
        #pragma unroll
        for (int q = 0; q < 8; ++q) sB[(c0 + q) * LDB_P + kp] = tmp[q];
      }
    }
    async_wait0();
    __syncthreads();

    if (kt + 32 < Kdim) {  // hint next tiles (global_prefetch_b8)
      __builtin_prefetch(A + (size_t)(m0 + (tid & 127)) * lda + kt + 32, 0, 1);
      if (TRANSB)
        __builtin_prefetch(Bsrc + (size_t)(n0 + (tid & 127)) * ldbs + kt + 32, 0, 1);
      else
        __builtin_prefetch(Bsrc + (size_t)(kt + 32 + (tid & 31)) * ldbs + n0, 0, 1);
    }

    // ---- fragments (contiguous b128 pairs) + WMMA ----
    v16bf af[4];
    #pragma unroll
    for (int mi = 0; mi < 4; ++mi)
      af[mi] = load_frag(sA + (wm * 64 + mi * 16 + (lane & 15)) * LDA_P + (lane & 16));
    v16bf bfm[2];
    #pragma unroll
    for (int ni = 0; ni < 2; ++ni)
      bfm[ni] = load_frag(sB + (wn * 32 + ni * 16 + (lane & 15)) * LDB_P + (lane & 16));
    #pragma unroll
    for (int mi = 0; mi < 4; ++mi)
      #pragma unroll
      for (int ni = 0; ni < 2; ++ni)
        acc[mi][ni] = __builtin_amdgcn_wmma_f32_16x16x32_bf16(
            false, af[mi], false, bfm[ni], (short)0, acc[mi][ni], false, false);
    __syncthreads();
  }
}

// ---------------- prep kernels ----------------

__global__ __launch_bounds__(256) void cvt_bf16_kernel(const float* __restrict__ in,
                                                       __bf16* __restrict__ out, int n) {
  int i = blockIdx.x * 256 + threadIdx.x;
  if (i < n) out[i] = f2bf(in[i]);
}

// [nmat][D][D] (d,e) -> bf16 [nmat][D][D] (e,d) so projection GEMMs are TRANSB.
__global__ __launch_bounds__(256) void cvt_transpose_kernel(const float* __restrict__ in,
                                                            __bf16* __restrict__ out,
                                                            int nmat) {
  int i = blockIdx.x * 256 + threadIdx.x;
  if (i < nmat * DD * DD) {
    int m = i / (DD * DD);
    int r = (i / DD) % DD;  // e
    int c = i % DD;         // d
    out[i] = f2bf(in[(size_t)m * DD * DD + (size_t)c * DD + r]);
  }
}

// Pack Whh [2][3H][H] into WMMA fragment order:
// whhP[(((dir*8+kc)*48+nt)*32+lane)*16+e] = Whh[dir][nt*16+(lane&15)][kc*32+frag_koff(lane,e)]
__global__ __launch_bounds__(256) void whh_pack_kernel(const float* __restrict__ whh,
                                                       __bf16* __restrict__ whhP) {
  int i = blockIdx.x * 256 + threadIdx.x;
  if (i >= 2 * 8 * 48 * 32 * 16) return;
  int e = i & 15;
  int t = i >> 4;
  int lane = t & 31; t >>= 5;
  int nt = t % 48; t /= 48;
  int kc = t & 7;
  int dir = t >> 3;
  int n = nt * 16 + (lane & 15);
  int k = kc * 32 + frag_koff(lane, e);
  whhP[i] = f2bf(whh[((size_t)dir * H3 + n) * HH + k]);
}

// am = emb[label], qmask = sign(|sum_d am|), bf16 copies of am and seq.
__global__ __launch_bounds__(256) void embed_kernel(const int* __restrict__ labels,
                                                    const float* __restrict__ seq,
                                                    const float* __restrict__ emb,
                                                    float* __restrict__ am,
                                                    __bf16* __restrict__ am_bf,
                                                    __bf16* __restrict__ seq_bf,
                                                    float* __restrict__ qmask) {
  __shared__ float red[256];
  int bs = blockIdx.x;
  int lbl = labels[bs];
  const float* row = emb + (size_t)lbl * DD;
  float part = 0.f;
  for (int j = threadIdx.x; j < DD; j += 256) {
    float v = row[j];
    am[(size_t)bs * DD + j]     = v;
    am_bf[(size_t)bs * DD + j]  = f2bf(v);
    seq_bf[(size_t)bs * DD + j] = f2bf(seq[(size_t)bs * DD + j]);
    part += v;
  }
  red[threadIdx.x] = part;
  __syncthreads();
  for (int s = 128; s > 0; s >>= 1) {
    if (threadIdx.x < s) red[threadIdx.x] += red[threadIdx.x + s];
    __syncthreads();
  }
  if (threadIdx.x == 0) qmask[bs] = (fabsf(red[0]) > 0.f) ? 1.f : 0.f;
}

// ---------------- GEMM kernels ----------------

// Q/K/V projections with pre-transposed weights (TRANSB, fully async staging).
__global__ __launch_bounds__(256) void gemm_qkv_kernel(
    const __bf16* __restrict__ am_bf, const __bf16* __restrict__ seq_bf,
    const __bf16* __restrict__ wqT, const __bf16* __restrict__ wkT, const __bf16* __restrict__ wvT,
    const float* __restrict__ bq, const float* __restrict__ bk, const float* __restrict__ bv,
    __bf16* __restrict__ Q, __bf16* __restrict__ K, __bf16* __restrict__ V) {
  __shared__ __bf16 sA[128 * LDA_P];
  __shared__ __bf16 sB[128 * LDB_P];
  int bat = blockIdx.y;
  int p = bat / NATT, ka = bat % NATT;
  const __bf16* A    = (p == 0) ? am_bf : seq_bf;
  const __bf16* W    = ((p == 0) ? wqT : (p == 1) ? wkT : wvT) + (size_t)ka * DD * DD;
  const float*  bias = ((p == 0) ? bq : (p == 1) ? bk : bv) + (size_t)ka * DD;
  __bf16*       Out  = ((p == 0) ? Q : (p == 1) ? K : V) + (size_t)ka * MQKV * DD;
  int tm = blockIdx.x / 6, tn = blockIdx.x % 6;
  int m0 = tm * 128, n0 = tn * 128;
  v8f acc[4][2];
  v8f zv = {0.f, 0.f, 0.f, 0.f, 0.f, 0.f, 0.f, 0.f};
  #pragma unroll
  for (int i = 0; i < 4; ++i)
    #pragma unroll
    for (int j = 0; j < 2; ++j) acc[i][j] = zv;
  gemm_block_128x128<true>(A, DD, m0, W, DD, n0, DD, sA, sB, acc);
  int lane = threadIdx.x & 31, wave = threadIdx.x >> 5, wm = wave & 1, wn = wave >> 1;
  #pragma unroll
  for (int mi = 0; mi < 4; ++mi)
    #pragma unroll
    for (int ni = 0; ni < 2; ++ni)
      #pragma unroll
      for (int r = 0; r < 8; ++r) {
        int gm = m0 + wm * 64 + mi * 16 + r + ((lane & 16) ? 8 : 0);
        int gn = n0 + wn * 32 + ni * 16 + (lane & 15);
        float v = acc[mi][ni][r] + bias[gn];
        Out[(size_t)gm * DD + gn] = f2bf(fmaxf(v, 0.f));
      }
}

__global__ __launch_bounds__(256) void gemm_scores_kernel(
    const __bf16* __restrict__ Q, const __bf16* __restrict__ Kbuf,
    float* __restrict__ scores) {
  __shared__ __bf16 sA[128 * LDA_P];
  __shared__ __bf16 sB[128 * LDB_P];
  int kb = blockIdx.y;
  const __bf16* A = Q    + (size_t)kb * SS * DD;
  const __bf16* B = Kbuf + (size_t)kb * SS * DD;
  int tm = blockIdx.x / 4, tn = blockIdx.x % 4;
  int m0 = tm * 128, n0 = tn * 128;
  v8f acc[4][2];
  v8f zv = {0.f, 0.f, 0.f, 0.f, 0.f, 0.f, 0.f, 0.f};
  #pragma unroll
  for (int i = 0; i < 4; ++i)
    #pragma unroll
    for (int j = 0; j < 2; ++j) acc[i][j] = zv;
  gemm_block_128x128<true>(A, DD, m0, B, DD, n0, DD, sA, sB, acc);
  const float scale = 0.036084391824351615f;  // 1/sqrt(768)
  int lane = threadIdx.x & 31, wave = threadIdx.x >> 5, wm = wave & 1, wn = wave >> 1;
  #pragma unroll
  for (int mi = 0; mi < 4; ++mi)
    #pragma unroll
    for (int ni = 0; ni < 2; ++ni)
      #pragma unroll
      for (int r = 0; r < 8; ++r) {
        int gm = m0 + wm * 64 + mi * 16 + r + ((lane & 16) ? 8 : 0);
        int gn = n0 + wn * 32 + ni * 16 + (lane & 15);
        scores[((size_t)kb * SS + gm) * SS + gn] = acc[mi][ni][r] * scale;
      }
}

// row softmax over m (512), post-softmax query mask, bf16 out.
// Scores row staged global->LDS with async-to-LDS copy.
__global__ __launch_bounds__(128) void softmax_kernel(const float* __restrict__ scores,
                                                      const float* __restrict__ qmask,
                                                      __bf16* __restrict__ attn) {
  __shared__ float red[128];
  __shared__ float srow[512];
  int row = blockIdx.x;
  int kb = row >> 9, q = row & 511;
  int b = kb & 7;
  int tid = threadIdx.x;
  async_copy_b128(srow + tid * 4, scores + (size_t)row * SS + tid * 4);
  async_wait0();
  __syncthreads();
  float mx = -3.4e38f;
  #pragma unroll
  for (int c = 0; c < 4; ++c) mx = fmaxf(mx, srow[tid + c * 128]);
  red[tid] = mx; __syncthreads();
  for (int s = 64; s > 0; s >>= 1) {
    if (tid < s) red[tid] = fmaxf(red[tid], red[tid + s]);
    __syncthreads();
  }
  float m = red[0];
  __syncthreads();
  float ev[4]; float sum = 0.f;
  #pragma unroll
  for (int c = 0; c < 4; ++c) { ev[c] = __expf(srow[tid + c * 128] - m); sum += ev[c]; }
  red[tid] = sum; __syncthreads();
  for (int s = 64; s > 0; s >>= 1) {
    if (tid < s) red[tid] += red[tid + s];
    __syncthreads();
  }
  float inv = qmask[b * SS + q] / red[0];
  #pragma unroll
  for (int c = 0; c < 4; ++c) attn[(size_t)row * SS + tid + c * 128] = f2bf(ev[c] * inv);
}

__global__ __launch_bounds__(256) void gemm_out_kernel(
    const __bf16* __restrict__ attn, const __bf16* __restrict__ V,
    const float* __restrict__ am, const float* __restrict__ seq,
    __bf16* __restrict__ elout) {
  __shared__ __bf16 sA[128 * LDA_P];
  __shared__ __bf16 sB[128 * LDB_P];
  int kb = blockIdx.y;
  int b = kb & 7;
  const __bf16* A = attn + (size_t)kb * SS * SS;
  const __bf16* B = V    + (size_t)kb * SS * DD;
  int tm = blockIdx.x / 6, tn = blockIdx.x % 6;
  int m0 = tm * 128, n0 = tn * 128;
  v8f acc[4][2];
  v8f zv = {0.f, 0.f, 0.f, 0.f, 0.f, 0.f, 0.f, 0.f};
  #pragma unroll
  for (int i = 0; i < 4; ++i)
    #pragma unroll
    for (int j = 0; j < 2; ++j) acc[i][j] = zv;
  gemm_block_128x128<false>(A, SS, m0, B, DD, n0, SS, sA, sB, acc);
  int lane = threadIdx.x & 31, wave = threadIdx.x >> 5, wm = wave & 1, wn = wave >> 1;
  #pragma unroll
  for (int mi = 0; mi < 4; ++mi)
    #pragma unroll
    for (int ni = 0; ni < 2; ++ni)
      #pragma unroll
      for (int r = 0; r < 8; ++r) {
        int gm = m0 + wm * 64 + mi * 16 + r + ((lane & 16) ? 8 : 0);
        int gn = n0 + wn * 32 + ni * 16 + (lane & 15);
        size_t bsidx = ((size_t)b * SS + gm) * DD + gn;
        float v = acc[mi][ni][r] + am[bsidx] + seq[bsidx];
        elout[((size_t)kb * SS + gm) * DD + gn] = f2bf(v);
      }
}

__global__ __launch_bounds__(256) void gemm_xp_kernel(
    const __bf16* __restrict__ elout, const __bf16* __restrict__ wih_bf,
    const float* __restrict__ bih, __bf16* __restrict__ xp) {
  __shared__ __bf16 sA[128 * LDA_P];
  __shared__ __bf16 sB[128 * LDB_P];
  int dir = blockIdx.y;
  const __bf16* B = wih_bf + (size_t)dir * H3 * DD;  // [h][d] row-major -> TRANSB
  int tm = blockIdx.x / 6, tn = blockIdx.x % 6;
  int m0 = tm * 128, n0 = tn * 128;
  v8f acc[4][2];
  v8f zv = {0.f, 0.f, 0.f, 0.f, 0.f, 0.f, 0.f, 0.f};
  #pragma unroll
  for (int i = 0; i < 4; ++i)
    #pragma unroll
    for (int j = 0; j < 2; ++j) acc[i][j] = zv;
  gemm_block_128x128<true>(elout, DD, m0, B, DD, n0, DD, sA, sB, acc);
  int lane = threadIdx.x & 31, wave = threadIdx.x >> 5, wm = wave & 1, wn = wave >> 1;
  #pragma unroll
  for (int mi = 0; mi < 4; ++mi)
    #pragma unroll
    for (int ni = 0; ni < 2; ++ni)
      #pragma unroll
      for (int r = 0; r < 8; ++r) {
        int gm = m0 + wm * 64 + mi * 16 + r + ((lane & 16) ? 8 : 0);
        int gn = n0 + wn * 32 + ni * 16 + (lane & 15);
        int nseq = gm >> 9, s = gm & 511;
        float v = acc[mi][ni][r] + bih[(size_t)dir * H3 + gn];
        xp[(((size_t)dir * SS + s) * NSEQ + nseq) * H3 + gn] = f2bf(v);
      }
}

// ---------------- GRU recurrence ----------------
// grid (6 groups of 16 sequences, 2 dirs); 256 threads = 8 waves.
// gh = h(16x256) @ Whh^T(256x768) via WMMA; B pre-packed in fragment order
// (coalesced b128 pair per fragment); h bf16 copy in LDS fragment order.
#define HBP 272
__global__ __launch_bounds__(256) void gru_kernel(const __bf16* __restrict__ xp,
                                                  const __bf16* __restrict__ whhP,
                                                  const float* __restrict__ bhh,
                                                  float* __restrict__ hfin) {
  __shared__ float  h[16 * HH];
  __shared__ __bf16 hbf[16 * HBP];
  __shared__ float  gh[16 * H3];
  const int g = blockIdx.x, dir = blockIdx.y;
  const int tid = threadIdx.x, lane = tid & 31, wave = tid >> 5;
  const int nbase = g * 16;

  for (int e = tid; e < 16 * HH; e += 256) h[e] = 0.f;
  for (int e = tid; e < 16 * HBP; e += 256) hbf[e] = f2bf(0.f);
  __syncthreads();

  v8f zv = {0.f, 0.f, 0.f, 0.f, 0.f, 0.f, 0.f, 0.f};
  const float* bhd = bhh + (size_t)dir * H3;

  for (int t = 0; t < SS; ++t) {
    int s = dir ? (SS - 1 - t) : t;
    v16bf afr[8];
    #pragma unroll
    for (int kc = 0; kc < 8; ++kc)
      afr[kc] = load_frag(hbf + (lane & 15) * HBP + kc * 32 + (lane & 16));
    #pragma unroll 1
    for (int j = 0; j < 6; ++j) {
      int n0 = wave * 96 + j * 16;
      int nt = wave * 6 + j;
      v8f acc = zv;
      #pragma unroll
      for (int kc = 0; kc < 8; ++kc) {
        v16bf bfr = load_frag(
            whhP + ((((size_t)dir * 8 + kc) * 48 + nt) * 32 + lane) * 16);
        acc = __builtin_amdgcn_wmma_f32_16x16x32_bf16(false, afr[kc], false, bfr,
                                                      (short)0, acc, false, false);
      }
      int mrow = (lane & 16) ? 8 : 0;
      #pragma unroll
      for (int r = 0; r < 8; ++r)
        gh[(r + mrow) * H3 + n0 + (lane & 15)] = acc[r];
    }
    __syncthreads();
    for (int e = tid; e < 16 * HH; e += 256) {
      int row = e >> 8, jj = e & 255;
      int n = nbase + row;
      if (n < NSEQ) {
        const __bf16* xprow = xp + (((size_t)dir * SS + s) * NSEQ + n) * H3;
        float ir  = bf2f(xprow[jj]);
        float iz  = bf2f(xprow[HH + jj]);
        float in_ = bf2f(xprow[2 * HH + jj]);
        float ghr = gh[row * H3 + jj] + bhd[jj];
        float ghz = gh[row * H3 + HH + jj] + bhd[HH + jj];
        float ghn = gh[row * H3 + 2 * HH + jj] + bhd[2 * HH + jj];
        float rg = 1.f / (1.f + __expf(-(ir + ghr)));
        float zg = 1.f / (1.f + __expf(-(iz + ghz)));
        float ng = tanhf(in_ + rg * ghn);
        float hnew = (1.f - zg) * ng + zg * h[row * HH + jj];
        h[row * HH + jj] = hnew;
        hbf[row * HBP + (jj & 224) + kperm(jj & 31)] = f2bf(hnew);
      }
    }
    __syncthreads();
  }
  for (int e = tid; e < 16 * HH; e += 256) {
    int row = e >> 8, jj = e & 255;
    int n = nbase + row;
    if (n < NSEQ) hfin[((size_t)dir * NSEQ + n) * HH + jj] = h[row * HH + jj];
  }
}

// logits[b,k] = sigmoid(concat(h_f[n], h_b[n]) . W1 + b1), n = k*8 + b
__global__ __launch_bounds__(64) void final_kernel(const float* __restrict__ hfin,
                                                   const float* __restrict__ W1,
                                                   const float* __restrict__ b1,
                                                   float* __restrict__ out) {
  __shared__ float red[64];
  int n = blockIdx.x, tid = threadIdx.x;
  float part = 0.f;
  for (int j = tid; j < 2 * HH; j += 64) {
    float hv = (j < HH) ? hfin[(size_t)n * HH + j]
                        : hfin[((size_t)NSEQ + n) * HH + (j - HH)];
    part += hv * W1[j];
  }
  red[tid] = part; __syncthreads();
  for (int s = 32; s > 0; s >>= 1) {
    if (tid < s) red[tid] += red[tid + s];
    __syncthreads();
  }
  if (tid == 0) {
    int k = n >> 3, b = n & 7;
    out[b * NATT + k] = 1.f / (1.f + __expf(-(red[0] + b1[0])));
  }
}

// ---------------- host launch ----------------

extern "C" void kernel_launch(void* const* d_in, const int* in_sizes, int n_in,
                              void* d_out, int out_size, void* d_ws, size_t ws_size,
                              hipStream_t stream) {
  const int*   labels = (const int*)d_in[0];
  const float* seq    = (const float*)d_in[1];
  const float* emb    = (const float*)d_in[2];
  const float* Wq     = (const float*)d_in[3];
  const float* bq     = (const float*)d_in[4];
  const float* Wk     = (const float*)d_in[5];
  const float* bk     = (const float*)d_in[6];
  const float* Wv     = (const float*)d_in[7];
  const float* bv     = (const float*)d_in[8];
  const float* gWih   = (const float*)d_in[9];
  const float* gWhh   = (const float*)d_in[10];
  const float* gbih   = (const float*)d_in[11];
  const float* gbhh   = (const float*)d_in[12];
  const float* W1     = (const float*)d_in[13];
  const float* b1     = (const float*)d_in[14];
  float* out = (float*)d_out;

  char* base = (char*)d_ws;
  size_t off = 0;
  auto alloc = [&](size_t bytes) -> void* {
    off = (off + 255) & ~(size_t)255;
    void* p = base + off;
    off += bytes;
    return p;
  };
  float*  am_f    = (float*)alloc((size_t)MQKV * DD * 4);
  float*  qmask   = (float*)alloc((size_t)MQKV * 4);
  __bf16* am_bf   = (__bf16*)alloc((size_t)MQKV * DD * 2);
  __bf16* seq_bf  = (__bf16*)alloc((size_t)MQKV * DD * 2);
  __bf16* wqT     = (__bf16*)alloc((size_t)NATT * DD * DD * 2);
  __bf16* wkT     = (__bf16*)alloc((size_t)NATT * DD * DD * 2);
  __bf16* wvT     = (__bf16*)alloc((size_t)NATT * DD * DD * 2);
  __bf16* wih_bf  = (__bf16*)alloc((size_t)2 * H3 * DD * 2);
  __bf16* whhP    = (__bf16*)alloc((size_t)2 * HH * H3 * 2);
  __bf16* Qb      = (__bf16*)alloc((size_t)NATT * MQKV * DD * 2);
  __bf16* Kb      = (__bf16*)alloc((size_t)NATT * MQKV * DD * 2);
  __bf16* Vb      = (__bf16*)alloc((size_t)NATT * MQKV * DD * 2);
  float*  scoresF = (float*)alloc((size_t)NSEQ * SS * SS * 4);
  __bf16* attn_bf = (__bf16*)alloc((size_t)NSEQ * SS * SS * 2);
  __bf16* elout   = (__bf16*)alloc((size_t)NSEQ * SS * DD * 2);
  __bf16* xp_bf   = (__bf16*)alloc((size_t)2 * SS * NSEQ * H3 * 2);
  float*  hfin    = (float*)alloc((size_t)2 * NSEQ * HH * 4);
  (void)ws_size; (void)n_in; (void)in_sizes; (void)out_size;

  int nW = NATT * DD * DD;
  cvt_transpose_kernel<<<(nW + 255) / 256, 256, 0, stream>>>(Wq, wqT, NATT);
  cvt_transpose_kernel<<<(nW + 255) / 256, 256, 0, stream>>>(Wk, wkT, NATT);
  cvt_transpose_kernel<<<(nW + 255) / 256, 256, 0, stream>>>(Wv, wvT, NATT);
  int nI = 2 * H3 * DD;
  cvt_bf16_kernel<<<(nI + 255) / 256, 256, 0, stream>>>(gWih, wih_bf, nI);
  int nT = 2 * 8 * 48 * 32 * 16;
  whh_pack_kernel<<<(nT + 255) / 256, 256, 0, stream>>>(gWhh, whhP);

  embed_kernel<<<MQKV, 256, 0, stream>>>(labels, seq, emb, am_f, am_bf, seq_bf, qmask);

  gemm_qkv_kernel<<<dim3(32 * 6, 33), 256, 0, stream>>>(am_bf, seq_bf, wqT, wkT, wvT,
                                                        bq, bk, bv, Qb, Kb, Vb);
  gemm_scores_kernel<<<dim3(4 * 4, NSEQ), 256, 0, stream>>>(Qb, Kb, scoresF);
  softmax_kernel<<<NSEQ * SS, 128, 0, stream>>>(scoresF, qmask, attn_bf);
  gemm_out_kernel<<<dim3(4 * 6, NSEQ), 256, 0, stream>>>(attn_bf, Vb, am_f, seq, elout);
  gemm_xp_kernel<<<dim3((MXP / 128) * 6, 2), 256, 0, stream>>>(elout, wih_bf, gbih, xp_bf);
  gru_kernel<<<dim3(6, 2), 256, 0, stream>>>(xp_bf, whhP, gbhh, hfin);
  final_kernel<<<NSEQ, 64, 0, stream>>>(hfin, W1, b1, out);
}
